// SupernodePoolingPosonly_72043781423642
// MI455X (gfx1250) — matrix operand order repaired and checked by reading, not compiled, not run on hardware
//
#include <hip/hip_runtime.h>
#include <cstdint>

typedef __attribute__((ext_vector_type(16))) _Float16 v16h;
typedef __attribute__((ext_vector_type(8)))  float    v8f;

#define HID   256
#define DEG   32
#define SUPW  4              // supernodes per workgroup (fused kernel)
#define ROWS  (SUPW * DEG)   // 128 edge rows per workgroup
#define NT256 16             // 256 / 16 N-tiles

// ln(10000)/32 and ln(10000)/42
#define OMEGA_K4 0.28782313662425572f
#define OMEGA_K3 0.21929381837953772f

union AFrag { v16h v; uint32_t d[8]; };
union BFrag { v16h v; uint4 q[2]; };

// Branch-free tanh-form GELU: one v_exp_f32 + one v_rcp_f32 + FMAs.
// tanh saturates correctly at +/-inf through the exp/rcp limits.
__device__ __forceinline__ float gelu_fast(float x) {
  float u  = 0.7978845608028654f * (x + 0.044715f * x * x * x);
  float e  = __expf(2.0f * u);
  float th = 1.0f - 2.0f * __builtin_amdgcn_rcpf(e + 1.0f);  // tanh(u)
  return 0.5f * x * (1.0f + th);
}

// ---------------------------------------------------------------------------
// Pack a row-major f32 weight (K x N) into f16 WMMA B-fragment order:
// dst[((kt*NT + nt)*32 + lane)*16 + hi] = W[kt*32 + (lane>>4)*8 + (hi&7) + (hi>>3)*16][nt*16 + (lane&15)]
// so each lane's 16 halves are contiguous (2 x b128 loads per fragment).
// ---------------------------------------------------------------------------
__global__ __launch_bounds__(256) void pack_b_kernel(
    const float* __restrict__ src, _Float16* __restrict__ dst, int K, int N) {
  int tid = blockIdx.x * blockDim.x + threadIdx.x;
  if (tid >= K * N) return;
  int hi   = tid & 15;
  int lane = (tid >> 4) & 31;
  int tile = tid >> 9;
  int NT   = N >> 4;
  int nt   = tile % NT;
  int kt   = tile / NT;
  int k = kt * 32 + ((lane >> 4) << 3) + (hi & 7) + ((hi >> 3) << 4);
  int n = nt * 16 + (lane & 15);
  dst[tid] = (_Float16)src[k * N + n];
}

// ---------------------------------------------------------------------------
// One 128x256x256 GEMM layer on LDS activations, packed-f16 weights from L2.
// Wave wv owns N-tiles {2wv, 2wv+1}; B fragments for all 8 K-steps held in
// registers; loops 8 M-tiles gathering A fragments from LDS.
// ---------------------------------------------------------------------------
__device__ __forceinline__ void gemm_layer(
    const _Float16* __restrict__ A_lds, const _Float16* __restrict__ Bp,
    const float* __restrict__ bias, _Float16* __restrict__ D_lds,
    bool do_gelu, int wv, int l) {
  const int col    = l & 15;
  const int loff_a = (l >> 4) * 8;
  for (int ni = 0; ni < 2; ++ni) {
    const int nt = wv * 2 + ni;
    BFrag b[8];
#pragma unroll
    for (int kt = 0; kt < 8; ++kt) {
      const uint4* p = reinterpret_cast<const uint4*>(Bp + (size_t)((kt * NT256 + nt) * 32 + l) * 16);
      b[kt].q[0] = p[0];
      b[kt].q[1] = p[1];
    }
    const int n    = nt * 16 + col;
    const float bn = bias[n];
    for (int mt = 0; mt < ROWS / 16; ++mt) {
      v8f c = {};
      const int rowA = mt * 16 + col;
      const _Float16* arow = A_lds + rowA * HID;
#pragma unroll
      for (int kt = 0; kt < 8; ++kt) {
        AFrag a;
        const uint32_t* ap = reinterpret_cast<const uint32_t*>(arow + kt * 32 + loff_a);
        a.d[0] = ap[0]; a.d[1] = ap[1]; a.d[2] = ap[2];  a.d[3] = ap[3];
        a.d[4] = ap[8]; a.d[5] = ap[9]; a.d[6] = ap[10]; a.d[7] = ap[11];
        c = __builtin_amdgcn_wmma_f32_16x16x32_f16(false, a.v, false, b[kt].v,
                                                   (short)0, c, false, false);
      }
      const int rbase = mt * 16 + ((l >> 4) * 8);
#pragma unroll
      for (int r = 0; r < 8; ++r) {
        float x = c[r] + bn;
        if (do_gelu) x = gelu_fast(x);
        D_lds[(rbase + r) * HID + n] = (_Float16)x;
      }
    }
  }
}

// ---------------------------------------------------------------------------
// Fused: relpos -> sincos embed -> Linear+GELU -> Linear -> segment mean,
// plus supernode abspos sincos embed; writes the 512-wide concat rows (f16).
// ---------------------------------------------------------------------------
__global__ __launch_bounds__(256) void edge_mlp_kernel(
    const float* __restrict__ pos, const int* __restrict__ sup_idx,
    const int* __restrict__ src_idx,
    const _Float16* __restrict__ w1p, const float* __restrict__ b1,
    const _Float16* __restrict__ w2p, const float* __restrict__ b2,
    _Float16* __restrict__ conc) {
  __shared__ _Float16 Xs[ROWS * HID];   // 64 KB: embed, later layer-2 output
  __shared__ _Float16 Hs[ROWS * HID];   // 64 KB: GELU(layer-1) activations
  __shared__ float spos[SUPW][3];
  __shared__ float om4[32];             // 10000^(-j/32)
  __shared__ float om3[42];             // 10000^(-j/42)

  const int t     = threadIdx.x;
  const int wv    = t >> 5;
  const int l     = t & 31;
  const int sBase = blockIdx.x * SUPW;

  if (t < SUPW * 3) {
    int g = t / 3, d = t % 3;
    spos[g][d] = pos[(size_t)sup_idx[sBase + g] * 3 + d];
  } else if (t >= 32 && t < 64) {
    om4[t - 32] = __expf(-OMEGA_K4 * (float)(t - 32));
  } else if (t >= 64 && t < 106) {
    om3[t - 64] = __expf(-OMEGA_K3 * (float)(t - 64));
  }
  __syncthreads();

  // ---- sincos relpos embed (ndim=4, eff=64): 2 threads per edge row ----
  {
    const int r  = t >> 1;
    const int c0 = (t & 1) * 128;
    const int e  = blockIdx.x * ROWS + r;
    const int si = src_idx[e];
    const float dx = spos[r >> 5][0] - pos[(size_t)si * 3 + 0];
    const float dy = spos[r >> 5][1] - pos[(size_t)si * 3 + 1];
    const float dz = spos[r >> 5][2] - pos[(size_t)si * 3 + 2];
    const float mg = sqrtf(dx * dx + dy * dy + dz * dz);
    const float rel[4] = {dx, dy, dz, mg};
    for (int c = c0; c < c0 + 128; ++c) {
      const int d = c >> 6;
      const int j = c & 63;
      float v;
      if (j < 32) v = __sinf(rel[d] * om4[j]);
      else        v = __cosf(rel[d] * om4[j - 32]);
      Xs[r * HID + c] = (_Float16)v;
    }
  }
  __syncthreads();

  gemm_layer(Xs, w1p, b1, Hs, true,  wv, l);   // Linear1 + GELU
  __syncthreads();
  gemm_layer(Hs, w2p, b2, Xs, false, wv, l);   // Linear2 (+b2) -> Xs
  __syncthreads();

  // ---- segment mean (each supernode has exactly DEG=32 edges) ----
  for (int g = 0; g < SUPW; ++g) {
    float s = 0.f;
#pragma unroll 8
    for (int i = 0; i < DEG; ++i) s += (float)Xs[(g * DEG + i) * HID + t];
    conc[(size_t)(sBase + g) * (2 * HID) + t] = (_Float16)(s * (1.0f / DEG));
  }
  // ---- supernode abspos embed (ndim=3, eff=84, 4 zero-pad cols) ----
  for (int g = 0; g < SUPW; ++g) {
    float v = 0.f;
    if (t < 252) {
      const int d = t / 84, j = t % 84;
      if (j < 42) v = __sinf(spos[g][d] * om3[j]);
      else        v = __cosf(spos[g][d] * om3[j - 42]);
    }
    conc[(size_t)(sBase + g) * (2 * HID) + HID + t] = (_Float16)v;
  }
}

// ---------------------------------------------------------------------------
// Projection GEMM: out(8192x256,f32) = conc(8192x512,f16) @ wp + bp.
// 32 rows per WG; wave wv: mt = wv&1, nt = 4*(wv>>1)+i; K = 16 steps of 32.
// ---------------------------------------------------------------------------
__global__ __launch_bounds__(256) void proj_gemm_kernel(
    const _Float16* __restrict__ conc, const _Float16* __restrict__ wpp,
    const float* __restrict__ bp, float* __restrict__ out) {
  const int t      = threadIdx.x;
  const int wv     = t >> 5;
  const int l      = t & 31;
  const int col    = l & 15;
  const int loff_a = (l >> 4) * 8;
  const int mBase  = blockIdx.x * 32;
  const int mt     = wv & 1;
  const int rowA   = mBase + mt * 16 + col;
  const _Float16* arow = conc + (size_t)rowA * 512;

  for (int i = 0; i < 4; ++i) {
    const int nt = (wv >> 1) * 4 + i;
    v8f c = {};
#pragma unroll
    for (int kt = 0; kt < 16; ++kt) {
      AFrag a;
      const uint32_t* ap = reinterpret_cast<const uint32_t*>(arow + kt * 32 + loff_a);
      a.d[0] = ap[0]; a.d[1] = ap[1]; a.d[2] = ap[2];  a.d[3] = ap[3];
      a.d[4] = ap[8]; a.d[5] = ap[9]; a.d[6] = ap[10]; a.d[7] = ap[11];
      BFrag b;
      const uint4* bptr = reinterpret_cast<const uint4*>(wpp + (size_t)((kt * NT256 + nt) * 32 + l) * 16);
      b.q[0] = bptr[0];
      b.q[1] = bptr[1];
      c = __builtin_amdgcn_wmma_f32_16x16x32_f16(false, a.v, false, b.v,
                                                 (short)0, c, false, false);
    }
    const int rbase = mBase + mt * 16 + ((l >> 4) * 8);
    const int n     = nt * 16 + col;
    const float bn  = bp[n];
#pragma unroll
    for (int r = 0; r < 8; ++r) out[(size_t)(rbase + r) * HID + n] = c[r] + bn;
  }
}

// ---------------------------------------------------------------------------
extern "C" void kernel_launch(void* const* d_in, const int* in_sizes, int n_in,
                              void* d_out, int out_size, void* d_ws, size_t ws_size,
                              hipStream_t stream) {
  const float* pos = (const float*)d_in[0];
  const int*   sup = (const int*)d_in[1];
  const int*   src = (const int*)d_in[2];
  // d_in[3] = dst_seg (unused; segmentation is implicit: DEG edges per supernode)
  const float* w1  = (const float*)d_in[4];
  const float* b1  = (const float*)d_in[5];
  const float* w2  = (const float*)d_in[6];
  const float* b2  = (const float*)d_in[7];
  const float* wp  = (const float*)d_in[8];
  const float* bp  = (const float*)d_in[9];
  float*       out = (float*)d_out;

  const int num_sup = in_sizes[1];          // 8192
  char* ws = (char*)d_ws;
  _Float16* w1p  = (_Float16*)(ws + 0);                     // 256*256*2 = 128 KB
  _Float16* w2p  = (_Float16*)(ws + (131072));              // 128 KB
  _Float16* wpp  = (_Float16*)(ws + (131072 * 2));          // 512*256*2 = 256 KB
  _Float16* conc = (_Float16*)(ws + (131072 * 2 + 262144)); // 8192*512*2 = 8 MB

  pack_b_kernel<<<(256 * 256 + 255) / 256, 256, 0, stream>>>(w1, w1p, 256, 256);
  pack_b_kernel<<<(256 * 256 + 255) / 256, 256, 0, stream>>>(w2, w2p, 256, 256);
  pack_b_kernel<<<(512 * 256 + 255) / 256, 256, 0, stream>>>(wp, wpp, 512, 256);

  edge_mlp_kernel<<<num_sup / SUPW, 256, 0, stream>>>(pos, sup, src,
                                                      w1p, b1, w2p, b2, conc);
  proj_gemm_kernel<<<num_sup / 32, 256, 0, stream>>>(conc, wpp, bp, out);
}
